// WindowRNN_67508295958746
// MI455X (gfx1250) — matrix operand order repaired and checked by reading, not compiled, not run on hardware
//
#include <hip/hip_runtime.h>
#include <cstdint>

#define H    256
#define BB   8
#define SS   2048
#define WIN  16
#define BM   64            // rows (positions) per block
#define LDSW 512           // halves per LDS row (x: 0..255, h: 256..511)
#define NBLK ((BB * SS) / BM)

typedef __attribute__((ext_vector_type(16))) __bf16 bf16x16;
typedef __attribute__((ext_vector_type(8)))  float  f32x8;

union Frag { bf16x16 v; uint4 q[2]; };

__device__ __forceinline__ unsigned f2bf1(float f) {
  union { float f; unsigned u; } c; c.f = f;
  return (c.u + 0x7FFFu + ((c.u >> 16) & 1u)) >> 16;   // RNE fp32 -> bf16
}
__device__ __forceinline__ unsigned pk2(float lo, float hi) {
  return f2bf1(lo) | (f2bf1(hi) << 16);
}
__device__ __forceinline__ float bf2f(unsigned h16) {
  union { unsigned u; float f; } c; c.u = h16 << 16;
  return c.f;
}
// XOR swizzle of 16B chunks within a row: conflict-free ds_b128 for 16 row-lanes
__device__ __forceinline__ int swz(int row, int col) {
  return row * LDSW + ((((col >> 3) ^ (row & 15)) << 3) | (col & 7));
}

// Hardware tanh on gfx1250 (TRANS32 op, co-issues with WMMA); branchless
// exp2/rcp fallback otherwise (no EXEC divergence in the epilogue).
__device__ __forceinline__ float fast_tanh(float v) {
#if __has_builtin(__builtin_amdgcn_tanhf)
  return __builtin_amdgcn_tanhf(v);
#else
  float a = __builtin_fabsf(v);
  float e = __builtin_amdgcn_exp2f(a * -2.8853900817779268f);   // exp(-2a)
  float r = 1.0f - 2.0f * e * __builtin_amdgcn_rcpf(1.0f + e);
  return __builtin_copysignf(r, v);
#endif
}

// ---------------------------------------------------------------------------
// Pre-kernel: pack Bcat = [W_ih^T ; W_hh^T] (512 x 256) into bf16 B-fragments.
// Fragment f = kk*16 + ntile; per lane 32B: column n = ntile*16 + (lane&15),
// K = kk*32 + (lane<16 ? 0 : 16) + {0..15}, 2 bf16 per dword (even K low).
// ---------------------------------------------------------------------------
__global__ __launch_bounds__(256) void wrnn_prep(const float* __restrict__ Wih,
                                                 const float* __restrict__ Whh,
                                                 unsigned short* __restrict__ wfrag) {
  int tid  = blockIdx.x * blockDim.x + threadIdx.x;   // 0..8191
  int lane = tid & 31;
  int f    = tid >> 5;                                // 0..255
  int nt   = f & 15;
  int kk   = f >> 4;
  int n    = nt * 16 + (lane & 15);
  int kb   = kk * 32 + ((lane & 16) ? 16 : 0);
  unsigned buf[8];
#pragma unroll
  for (int j = 0; j < 8; ++j) {
    int k0 = kb + 2 * j, k1 = k0 + 1;
    float w0 = (k0 < H) ? Wih[n * H + k0] : Whh[n * H + (k0 - H)];
    float w1 = (k1 < H) ? Wih[n * H + k1] : Whh[n * H + (k1 - H)];
    buf[j] = pk2(w0, w1);
  }
  uint4* dst = (uint4*)(wfrag + (size_t)f * 512 + lane * 16);
  dst[0] = *(uint4*)&buf[0];
  dst[1] = *(uint4*)&buf[4];
}

// ---------------------------------------------------------------------------
// Main kernel: windowed RNN, h carried as bf16 in LDS, 16x16x32 bf16 WMMA.
// ---------------------------------------------------------------------------
__global__ __launch_bounds__(256) void wrnn_main(const float* __restrict__ x,
                                                 const float* __restrict__ bih,
                                                 const float* __restrict__ bhh,
                                                 const unsigned short* __restrict__ wfrag,
                                                 float* __restrict__ out) {
  __shared__ alignas(16) unsigned short As[BM * LDSW];   // 64 KB: [x_k | h] bf16

  const int tid  = threadIdx.x;
  const int lane = tid & 31;
  const int wv   = tid >> 5;          // wave 0..7
  const int lm   = lane & 15;
  const int hi   = (lane >> 4) & 1;
  const int nbase   = wv * 2;         // this wave owns ntiles {nbase, nbase+1}, all 4 rowtiles
  const int rowbase = blockIdx.x * BM;

  // staging role: each thread owns one quarter-row (64 floats)
  const int srow = tid >> 2;          // 0..63
  const int sseg = tid & 3;           // 0..3
  const int p    = rowbase + srow;    // global position (b,t); BM | SS so b uniform per block
  const int bidx = p >> 11;
  const int tpos = p & (SS - 1);
  const float* xrow_base = x + (size_t)bidx * SS * H;

  // per-lane biases for this wave's two ntiles
  const int   n0    = nbase * 16 + lm;
  const float bias0 = bih[n0] + bhh[n0];
  const float bias1 = bih[n0 + 16] + bhh[n0 + 16];

  // zero the h region (cols 256..511)
#pragma unroll
  for (int c = 0; c < 8; ++c) {
    uint4 z = {0u, 0u, 0u, 0u};
    *(uint4*)&As[swz(srow, 256 + sseg * 64 + c * 8)] = z;
  }

#pragma unroll 1
  for (int k = 0; k < WIN; ++k) {
    // ---- stage x_k rows (bf16) into As[:, 0:256) ----
    int  tsrc  = tpos + k - (WIN - 1);          // left-padded window
    bool valid = (tsrc >= 0);
    const float4* xr = (const float4*)(xrow_base + (size_t)(valid ? tsrc : 0) * H);
#pragma unroll
    for (int c = 0; c < 8; ++c) {
      float4 v0 = {0.f, 0.f, 0.f, 0.f}, v1 = v0;
      if (valid) { v0 = xr[sseg * 16 + 2 * c]; v1 = xr[sseg * 16 + 2 * c + 1]; }
      uint4 pkd;
      pkd.x = pk2(v0.x, v0.y); pkd.y = pk2(v0.z, v0.w);
      pkd.z = pk2(v1.x, v1.y); pkd.w = pk2(v1.z, v1.w);
      *(uint4*)&As[swz(srow, sseg * 64 + c * 8)] = pkd;
    }
    __syncthreads();   // x_k and previous h visible to all waves

    // ---- GEMM: acc[rt][nt] += A(64x512) * Bcat(512x256) on this wave's tiles ----
    f32x8 acc[4][2];
#pragma unroll
    for (int rt = 0; rt < 4; ++rt)
#pragma unroll
      for (int nt = 0; nt < 2; ++nt) {
        f32x8 z = {0.f, 0.f, 0.f, 0.f, 0.f, 0.f, 0.f, 0.f};
        acc[rt][nt] = z;
      }

#pragma unroll 1
    for (int kk = 0; kk < 16; ++kk) {
      Frag A[4];
      int c0 = kk * 32 + hi * 16;                // A 16-bit 16x32 lane/K split (contiguous K = 16*hi + s)
#pragma unroll
      for (int rt = 0; rt < 4; ++rt) {
        int row = rt * 16 + lm;
        A[rt].q[0] = *(const uint4*)&As[swz(row, c0)];
        A[rt].q[1] = *(const uint4*)&As[swz(row, c0 + 8)];
      }
      Frag Bf[2];
#pragma unroll
      for (int nt = 0; nt < 2; ++nt) {
        const uint4* bp =
            (const uint4*)(wfrag + (size_t)(kk * 16 + nbase + nt) * 512 + lane * 16);
        Bf[nt].q[0] = bp[0];
        Bf[nt].q[1] = bp[1];
      }
#pragma unroll
      for (int rt = 0; rt < 4; ++rt)
#pragma unroll
        for (int nt = 0; nt < 2; ++nt)
          acc[rt][nt] = __builtin_amdgcn_wmma_f32_16x16x32_bf16(
              false, A[rt].v, false, Bf[nt].v, (short)0, acc[rt][nt], false, false);
    }

    __syncthreads();   // all A reads done before h (and next x) overwrite

    // ---- bias + hw tanh; write h back as bf16 A-rows (uniform, no branches) ----
#pragma unroll
    for (int rt = 0; rt < 4; ++rt) {
#pragma unroll
      for (int nt = 0; nt < 2; ++nt) {
        float bias = nt ? bias1 : bias0;
        int   ncol = (nbase + nt) * 16 + lm;
#pragma unroll
        for (int e = 0; e < 8; ++e) {
          int   row  = rt * 16 + e + hi * 8;     // C layout: M = e + 8*hi
          float hval = fast_tanh(acc[rt][nt][e] + bias);
          As[swz(row, 256 + ncol)] = (unsigned short)f2bf1(hval);
        }
      }
    }
  }

  // ---- final: read h_last from LDS (bf16), expand to f32, coalesced store ----
  __syncthreads();
  float* orow = out + (size_t)p * H;
#pragma unroll
  for (int c = 0; c < 8; ++c) {
    uint4 q = *(const uint4*)&As[swz(srow, 256 + sseg * 64 + c * 8)];
    float4 o0, o1;
    o0.x = bf2f(q.x & 0xFFFFu); o0.y = bf2f(q.x >> 16);
    o0.z = bf2f(q.y & 0xFFFFu); o0.w = bf2f(q.y >> 16);
    o1.x = bf2f(q.z & 0xFFFFu); o1.y = bf2f(q.z >> 16);
    o1.z = bf2f(q.w & 0xFFFFu); o1.w = bf2f(q.w >> 16);
    float4* dst = (float4*)(orow + sseg * 64 + c * 8);
    dst[0] = o0;
    dst[1] = o1;
  }
}

extern "C" void kernel_launch(void* const* d_in, const int* in_sizes, int n_in,
                              void* d_out, int out_size, void* d_ws, size_t ws_size,
                              hipStream_t stream) {
  const float* x   = (const float*)d_in[0];
  const float* Wih = (const float*)d_in[1];
  const float* Whh = (const float*)d_in[2];
  const float* bih = (const float*)d_in[3];
  const float* bhh = (const float*)d_in[4];
  unsigned short* wfrag = (unsigned short*)d_ws;   // needs 256 KB scratch

  wrnn_prep<<<32, 256, 0, stream>>>(Wih, Whh, wfrag);
  wrnn_main<<<NBLK, 256, 0, stream>>>(x, bih, bhh, wfrag, (float*)d_out);

  (void)in_sizes; (void)n_in; (void)out_size; (void)ws_size;
}
